// GCN_LSTM_90829968375902
// MI455X (gfx1250) — matrix operand order repaired and checked by reading, not compiled, not run on hardware
//
#include <hip/hip_runtime.h>
#include <hip/hip_bf16.h>
#include <math.h>

// ---------------------------------------------------------------------------
// GCN + LSTM + FC for MI455X (gfx1250), wave32 + WMMA f16->f32.
// ---------------------------------------------------------------------------

typedef __attribute__((ext_vector_type(16))) _Float16 v16h;
typedef __attribute__((ext_vector_type(8)))  _Float16 v8h;
typedef __attribute__((ext_vector_type(4)))  _Float16 v4h;
typedef __attribute__((ext_vector_type(8)))  float    v8f;

#define N_NODES 50000
#define N_EDGES 1600000
#define T_STEPS 9
#define HID 128

__device__ __forceinline__ void atomAddF32(float* p, float v) {
  unsafeAtomicAdd(p, v);   // lowers to global_atomic_add_f32 on gfx1250
}

__device__ __forceinline__ float sigmf(float x) { return 1.0f / (1.0f + expf(-x)); }

// Build a v16h fragment from two contiguous 16-byte chunks.
__device__ __forceinline__ v16h make_frag(const _Float16* p0, const _Float16* p1) {
  v8h lo = *(const v8h*)p0;
  v8h hi = *(const v8h*)p1;
  return __builtin_shufflevector(lo, hi, 0,1,2,3,4,5,6,7,8,9,10,11,12,13,14,15);
}

// ---------------------------------------------------------------------------
// Small elementwise / setup kernels
// ---------------------------------------------------------------------------

__global__ __launch_bounds__(256) void init_state_kernel(float* deg, float* c,
                                                         _Float16* hf16, int n) {
  int i = blockIdx.x * blockDim.x + threadIdx.x;
  int total = n * HID;
  if (i < total) { c[i] = 0.0f; hf16[i] = (_Float16)0.0f; }
  if (i < n) deg[i] = 1.0f;  // self-loop
}

__global__ __launch_bounds__(256) void deg_scatter_kernel(const long long* __restrict__ ei,
                                                          float* deg, int e_cnt) {
  int e = blockIdx.x * blockDim.x + threadIdx.x;
  if (e < e_cnt) atomAddF32(&deg[(int)ei[(size_t)e_cnt + e]], 1.0f);
}

__global__ __launch_bounds__(256) void dinv_kernel(float* deg, int n) {
  int i = blockIdx.x * blockDim.x + threadIdx.x;
  if (i < n) { float d = deg[i]; deg[i] = (d > 0.0f) ? rsqrtf(d) : 0.0f; }
}

__global__ __launch_bounds__(256) void norm_kernel(const long long* __restrict__ ei,
                                                   const float* __restrict__ dinv,
                                                   float* nrm, int e_cnt) {
  int e = blockIdx.x * blockDim.x + threadIdx.x;
  if (e < e_cnt) nrm[e] = dinv[(int)ei[e]] * dinv[(int)ei[(size_t)e_cnt + e]];
}

__global__ __launch_bounds__(256) void cvt_x_kernel(const float* __restrict__ x,
                                                    _Float16* __restrict__ xf, int total) {
  int i = blockIdx.x * blockDim.x + threadIdx.x;
  if (i < total) xf[i] = (_Float16)x[i];
}

__global__ __launch_bounds__(256) void cvt_w_kernel(const float* __restrict__ wgcn,
                                                    const float* __restrict__ wih,
                                                    const float* __restrict__ whh,
                                                    _Float16* __restrict__ wgcnT,
                                                    _Float16* __restrict__ wih16,
                                                    _Float16* __restrict__ whh16) {
  int i = blockIdx.x * blockDim.x + threadIdx.x;
  if (i < 512 * 128) { wih16[i] = (_Float16)wih[i]; whh16[i] = (_Float16)whh[i]; }
  if (i < 128 * 128) {
    int k = i >> 7, h = i & 127;
    wgcnT[h * 128 + k] = (_Float16)wgcn[i];   // column-contiguous B for WMMA
  }
}

__global__ __launch_bounds__(256) void agg_init_kernel(const float* __restrict__ xw,
                                                       const float* __restrict__ dinv,
                                                       float* __restrict__ agg, int total) {
  int i = blockIdx.x * blockDim.x + threadIdx.x;
  if (i < total) {
    int n = (i >> 7) % N_NODES;
    float dv = dinv[n];
    agg[i] = xw[i] * dv * dv;   // self-loop contribution
  }
}

__global__ __launch_bounds__(256) void relu_emb_kernel(const float* __restrict__ agg,
                                                       const float* __restrict__ bgcn,
                                                       _Float16* __restrict__ emb, int total) {
  int i = blockIdx.x * blockDim.x + threadIdx.x;
  if (i < total) {
    float v = agg[i] + bgcn[i & 127];
    emb[i] = (_Float16)fmaxf(v, 0.0f);
  }
}

// ---------------------------------------------------------------------------
// Edge scatter (one timestep): wave per edge, 4 floats per lane, L2-resident.
// ---------------------------------------------------------------------------
__global__ __launch_bounds__(256) void edge_scatter_kernel(const long long* __restrict__ ei,
                                                           const float* __restrict__ nrm,
                                                           const float* __restrict__ xw_t,
                                                           float* __restrict__ agg_t, int e_cnt) {
  int wave = threadIdx.x >> 5, lane = threadIdx.x & 31;
  int e = blockIdx.x * 8 + wave;
  if (e >= e_cnt) return;
  int s = (int)ei[e];
  int d = (int)ei[(size_t)e_cnt + e];
  float nm = nrm[e];
  const float4 v = *(const float4*)(xw_t + (size_t)s * HID + lane * 4);
  float* dp = agg_t + (size_t)d * HID + lane * 4;
  atomAddF32(dp + 0, v.x * nm);
  atomAddF32(dp + 1, v.y * nm);
  atomAddF32(dp + 2, v.z * nm);
  atomAddF32(dp + 3, v.w * nm);
}

// ---------------------------------------------------------------------------
// GEMM 1: xw[M=450000][128] = x_f16[M][128] @ WgcnT^T  (K=128), WMMA f16.
// Block = 8 waves; B (32 KB) staged in LDS; wave does 16 rows x 128 cols.
// ---------------------------------------------------------------------------
__global__ __launch_bounds__(256) void gemm_xw_kernel(const _Float16* __restrict__ xf16,
                                                      const _Float16* __restrict__ wgcnT,
                                                      float* __restrict__ xw, int m_rows) {
  __shared__ alignas(16) _Float16 ldsB[128 * 128];
  {
    const uint4* s = (const uint4*)wgcnT;
    uint4* d = (uint4*)ldsB;
    for (int i = threadIdx.x; i < 128 * 128 / 8; i += 256) d[i] = s[i];
  }
  __syncthreads();

  int wave = threadIdx.x >> 5, lane = threadIdx.x & 31;
  int g = lane >> 4, nl = lane & 15;
  int rt = blockIdx.x * 8 + wave;
  if (rt * 16 >= m_rows) return;
  size_t baseRow = (size_t)rt * 16;

  v8f acc[8] = {};
  const _Float16* arow = xf16 + (baseRow + nl) * HID;
#pragma unroll
  for (int kc = 0; kc < 4; ++kc) {
    v16h a = make_frag(arow + kc * 32 + g * 8, arow + kc * 32 + 16 + g * 8);
#pragma unroll
    for (int nt = 0; nt < 8; ++nt) {
      const _Float16* brow = ldsB + (nt * 16 + nl) * 128 + kc * 32 + g * 8;
      v16h b = make_frag(brow, brow + 16);
      acc[nt] = __builtin_amdgcn_wmma_f32_16x16x32_f16(false, a, false, b,
                                                       (short)0, acc[nt], false, false);
    }
  }
#pragma unroll
  for (int nt = 0; nt < 8; ++nt)
#pragma unroll
    for (int r = 0; r < 8; ++r)
      xw[(baseRow + r + 8 * g) * HID + nt * 16 + nl] = acc[nt][r];
}

// ---------------------------------------------------------------------------
// GEMM 2 (per timestep): gates[N][512] = emb_t @ Wih^T + h @ Whh^T + bias.
// grid.y = column group (64 cols); 64x128 f16 tiles of both weights in LDS.
// ---------------------------------------------------------------------------
__global__ __launch_bounds__(256) void gemm_gates_kernel(const _Float16* __restrict__ embt,
                                                         const _Float16* __restrict__ hf16,
                                                         const _Float16* __restrict__ wih,
                                                         const _Float16* __restrict__ whh,
                                                         const float* __restrict__ b_ih,
                                                         const float* __restrict__ b_hh,
                                                         float* __restrict__ gates, int n_rows) {
  __shared__ alignas(16) _Float16 ldsW[2 * 64 * 128];
  int colg = blockIdx.y;  // 0..7
  {
    const uint4* s0 = (const uint4*)(wih + (size_t)colg * 64 * 128);
    const uint4* s1 = (const uint4*)(whh + (size_t)colg * 64 * 128);
    uint4* d0 = (uint4*)ldsW;
    uint4* d1 = (uint4*)(ldsW + 64 * 128);
    for (int i = threadIdx.x; i < 64 * 128 / 8; i += 256) { d0[i] = s0[i]; d1[i] = s1[i]; }
  }
  __syncthreads();

  int wave = threadIdx.x >> 5, lane = threadIdx.x & 31;
  int g = lane >> 4, nl = lane & 15;
  int rt = blockIdx.x * 8 + wave;
  if (rt * 16 >= n_rows) return;
  size_t baseRow = (size_t)rt * 16;

  v8f acc[4] = {};
  const _Float16* ax = embt + (baseRow + nl) * HID;
  const _Float16* ah = hf16 + (baseRow + nl) * HID;
#pragma unroll
  for (int kc = 0; kc < 4; ++kc) {
    v16h a = make_frag(ax + kc * 32 + g * 8, ax + kc * 32 + 16 + g * 8);
#pragma unroll
    for (int nt = 0; nt < 4; ++nt) {
      const _Float16* brow = ldsW + (nt * 16 + nl) * 128 + kc * 32 + g * 8;
      v16h b = make_frag(brow, brow + 16);
      acc[nt] = __builtin_amdgcn_wmma_f32_16x16x32_f16(false, a, false, b,
                                                       (short)0, acc[nt], false, false);
    }
  }
#pragma unroll
  for (int kc = 0; kc < 4; ++kc) {
    v16h a = make_frag(ah + kc * 32 + g * 8, ah + kc * 32 + 16 + g * 8);
#pragma unroll
    for (int nt = 0; nt < 4; ++nt) {
      const _Float16* brow = ldsW + 64 * 128 + (nt * 16 + nl) * 128 + kc * 32 + g * 8;
      v16h b = make_frag(brow, brow + 16);
      acc[nt] = __builtin_amdgcn_wmma_f32_16x16x32_f16(false, a, false, b,
                                                       (short)0, acc[nt], false, false);
    }
  }
#pragma unroll
  for (int nt = 0; nt < 4; ++nt) {
    int col = colg * 64 + nt * 16 + nl;
    float bias = b_ih[col] + b_hh[col];
#pragma unroll
    for (int r = 0; r < 8; ++r)
      gates[(baseRow + r + 8 * g) * 512 + col] = acc[nt][r] + bias;
  }
}

// ---------------------------------------------------------------------------
// LSTM pointwise + fused final FC (wave per node; shuffle reduction for dot).
// ---------------------------------------------------------------------------
__global__ __launch_bounds__(256) void lstm_point_kernel(const float* __restrict__ gates,
                                                         float* __restrict__ cbuf,
                                                         _Float16* __restrict__ hf16,
                                                         const float* __restrict__ wfc,
                                                         const float* __restrict__ bfc,
                                                         float* __restrict__ out,
                                                         int n_rows, int t) {
  int wave = threadIdx.x >> 5, lane = threadIdx.x & 31;
  int n = blockIdx.x * 8 + wave;
  if (n >= n_rows) return;
  int j = lane * 4;
  const float* gr = gates + (size_t)n * 512;
  float4 gi = *(const float4*)(gr + j);
  float4 gf = *(const float4*)(gr + 128 + j);
  float4 gg = *(const float4*)(gr + 256 + j);
  float4 go = *(const float4*)(gr + 384 + j);
  float4 cp = *(float4*)(cbuf + (size_t)n * HID + j);

  float iv[4] = {gi.x, gi.y, gi.z, gi.w};
  float fv[4] = {gf.x, gf.y, gf.z, gf.w};
  float gv[4] = {gg.x, gg.y, gg.z, gg.w};
  float ov[4] = {go.x, go.y, go.z, go.w};
  float cv[4] = {cp.x, cp.y, cp.z, cp.w};
  float hv[4];
  float partial = 0.0f;
#pragma unroll
  for (int k = 0; k < 4; ++k) {
    float I = sigmf(iv[k]), F = sigmf(fv[k]), G = tanhf(gv[k]), O = sigmf(ov[k]);
    float C = F * cv[k] + I * G;
    float Hh = O * tanhf(C);
    cv[k] = C; hv[k] = Hh;
    partial += Hh * wfc[j + k];
  }
  *(float4*)(cbuf + (size_t)n * HID + j) = make_float4(cv[0], cv[1], cv[2], cv[3]);
  v4h hh; hh[0] = (_Float16)hv[0]; hh[1] = (_Float16)hv[1];
  hh[2] = (_Float16)hv[2]; hh[3] = (_Float16)hv[3];
  *(v4h*)(hf16 + (size_t)n * HID + j) = hh;

#pragma unroll
  for (int off = 16; off > 0; off >>= 1) partial += __shfl_down(partial, off, 32);
  if (lane == 0) out[(size_t)n * T_STEPS + t] = partial + bfc[0];
}

// ---------------------------------------------------------------------------
// Host launcher
// ---------------------------------------------------------------------------
extern "C" void kernel_launch(void* const* d_in, const int* in_sizes, int n_in,
                              void* d_out, int out_size, void* d_ws, size_t ws_size,
                              hipStream_t stream) {
  const float*     x_seq = (const float*)d_in[0];
  const long long* edge  = (const long long*)d_in[1];
  const float*     W_gcn = (const float*)d_in[2];
  const float*     b_gcn = (const float*)d_in[3];
  const float*     W_ih  = (const float*)d_in[4];
  const float*     W_hh  = (const float*)d_in[5];
  const float*     b_ih  = (const float*)d_in[6];
  const float*     b_hh  = (const float*)d_in[7];
  const float*     W_fc  = (const float*)d_in[8];
  const float*     b_fc  = (const float*)d_in[9];
  float* out = (float*)d_out;

  const int N = N_NODES, E = N_EDGES, T = T_STEPS;
  const size_t TOT = (size_t)T * N * HID;  // 57.6M elements

  char* ws = (char*)d_ws;
  size_t off = 0;
  auto carve = [&](size_t bytes) -> void* {
    void* p = ws + off;
    off += (bytes + 255) & ~(size_t)255;
    return p;
  };
  _Float16* xf16  = (_Float16*)carve(TOT * 2);          // 115.2 MB
  float*    xw    = (float*)   carve(TOT * 4);          // 230.4 MB
  float*    agg   = (float*)   carve(TOT * 4);          // 230.4 MB
  float*    gates = (float*)   carve((size_t)N * 512 * 4); // 102.4 MB
  float*    cbuf  = (float*)   carve((size_t)N * HID * 4);
  _Float16* hf16  = (_Float16*)carve((size_t)N * HID * 2);
  float*    deg   = (float*)   carve((size_t)N * 4);    // becomes dinv in place
  float*    nrm   = (float*)   carve((size_t)E * 4);
  _Float16* wgcnT = (_Float16*)carve(128 * 128 * 2);
  _Float16* wih16 = (_Float16*)carve(512 * 128 * 2);
  _Float16* whh16 = (_Float16*)carve(512 * 128 * 2);
  _Float16* emb   = (_Float16*)xw;  // alias: xw is dead after scatter completes

  int blkTot = (int)((TOT + 255) / 256);
  int blkNH  = (N * HID + 255) / 256;

  // 1) degree / normalization
  init_state_kernel<<<blkNH, 256, 0, stream>>>(deg, cbuf, hf16, N);
  deg_scatter_kernel<<<(E + 255) / 256, 256, 0, stream>>>(edge, deg, E);
  dinv_kernel<<<(N + 255) / 256, 256, 0, stream>>>(deg, N);
  norm_kernel<<<(E + 255) / 256, 256, 0, stream>>>(edge, deg, nrm, E);

  // 2) precision converts
  cvt_x_kernel<<<blkTot, 256, 0, stream>>>(x_seq, xf16, (int)TOT);
  cvt_w_kernel<<<(512 * 128 + 255) / 256, 256, 0, stream>>>(W_gcn, W_ih, W_hh,
                                                            wgcnT, wih16, whh16);

  // 3) xw = x @ W_gcn   (M = T*N = 450000 rows)
  {
    int m_rows = T * N;
    int tiles = m_rows / 16;            // 28125
    int blocks = (tiles + 7) / 8;       // 3516
    gemm_xw_kernel<<<blocks, 256, 0, stream>>>(xf16, wgcnT, xw, m_rows);
  }

  // 4) self-loop init + per-timestep scatter (keeps xw_t+agg_t in L2)
  agg_init_kernel<<<blkTot, 256, 0, stream>>>(xw, deg, agg, (int)TOT);
  for (int t = 0; t < T; ++t) {
    edge_scatter_kernel<<<(E + 7) / 8, 256, 0, stream>>>(
        edge, nrm, xw + (size_t)t * N * HID, agg + (size_t)t * N * HID, E);
  }

  // 5) relu + bias -> f16 embeddings (written over dead xw region)
  relu_emb_kernel<<<blkTot, 256, 0, stream>>>(agg, b_gcn, emb, (int)TOT);

  // 6) LSTM over time + fused FC
  {
    int tiles = N / 16;                 // 3125
    dim3 ggrid((tiles + 7) / 8, 8);     // (391, 8 col groups)
    int pblocks = (N + 7) / 8;          // 6250
    for (int t = 0; t < T; ++t) {
      gemm_gates_kernel<<<ggrid, 256, 0, stream>>>(
          emb + (size_t)t * N * HID, hf16, wih16, whh16, b_ih, b_hh, gates, N);
      lstm_point_kernel<<<pblocks, 256, 0, stream>>>(gates, cbuf, hf16,
                                                     W_fc, b_fc, out, N, t);
    }
  }
}
